// Edge_Discriminator_16827681865828
// MI455X (gfx1250) — matrix operand order repaired and checked by reading, not compiled, not run on hardware
//
#include <hip/hip_runtime.h>
#include <hip/hip_bf16.h>
#include <math.h>

#define N_NODES 100000
#define N_EDGES 1600000
#define IN_DIM  256
#define HID     64
#define EOSV    1e-10f
#define ALPHAV  0.1f
#define BIASV   1e-4f

typedef __attribute__((ext_vector_type(16))) __bf16 v16bf;
typedef __attribute__((ext_vector_type(8)))  float  v8f;

union BF16Frag {
    v16bf          v;
    unsigned short h[16];
    unsigned int   u[8];
};

// ---------------------------------------------------------------------------
// Kernel A: fused  q = relu(F @ W_emb + b_emb) @ (W_edge[:64] + W_edge[64:])
// Split-bf16 WMMA GEMM (hi*hi + hi*lo + lo*hi), fp32 accumulate.
// One wave = 16 nodes x 64 hidden. 8 K-steps of 32.
// ---------------------------------------------------------------------------
__launch_bounds__(256, 2)
__global__ void embed_gemm_kernel(const float* __restrict__ feat,
                                  const float* __restrict__ Wemb,
                                  const float* __restrict__ bemb,
                                  const float* __restrict__ Wedge,
                                  float* __restrict__ q)
{
    // B fragments, pre-swizzled: [hi/lo][ki(8)][tile(4)][jp(4)][lane(32)][4 halves]
    // Lane stride = 8B = 2 dwords -> ds_load_b64 reads hit all 64 banks once.
    __shared__ unsigned short ldsB[2][8][4][4][32][4];   // 64 KB

    // Cooperative fill: split W_emb (fp32) into bf16 hi + bf16 lo residual.
    for (int idx = threadIdx.x; idx < IN_DIM * HID; idx += blockDim.x) {
        int k = idx >> 6;            // 0..255
        int n = idx & 63;            // 0..63
        float w = Wemb[idx];         // row-major [k][n]
        unsigned bits = __float_as_uint(w);
        unsigned short hi16 = (unsigned short)(bits >> 16);
        float lo = w - __uint_as_float(bits & 0xFFFF0000u);
        unsigned short lo16 = (unsigned short)(__float_as_uint(lo) >> 16);
        int ki    = k >> 5;
        int krem  = k & 31;
        int khalf = (krem >> 3) & 1; // lane half owning this K (table order)
        int j     = (krem & 7) | ((krem >> 4) << 3);   // half index within fragment
        int t     = n >> 4;          // 16-column tile
        int L     = (khalf << 4) | (n & 15);   // owning lane
        ldsB[0][ki][t][j >> 2][L][j & 3] = hi16;
        ldsB[1][ki][t][j >> 2][L][j & 3] = lo16;
    }
    __syncthreads();

    const int lane = threadIdx.x & 31;
    const int wave = threadIdx.x >> 5;
    const int tile = blockIdx.x * 8 + wave;          // 16-node tile id
    if (tile * 16 >= N_NODES) return;                // wave-uniform exit

    const int row = tile * 16 + (lane & 15);
    const float* fp = feat + (size_t)row * IN_DIM;
    const int kA0 = (lane >> 4) << 3;                // 0 or 8 (A-matrix K phase)

    v8f acc[4] = {};                                 // 16x16 fp32 C tiles

#pragma unroll
    for (int ki = 0; ki < 8; ++ki) {
        const int k0 = ki * 32 + kA0;
        // A layout (16-bit 16x32): lane holds K = k0..k0+7 and k0+16..k0+23
        float4 a0 = *(const float4*)(fp + k0);
        float4 a1 = *(const float4*)(fp + k0 + 4);
        float4 b0 = *(const float4*)(fp + k0 + 16);
        float4 b1 = *(const float4*)(fp + k0 + 20);
        float af[16] = {a0.x, a0.y, a0.z, a0.w, a1.x, a1.y, a1.z, a1.w,
                        b0.x, b0.y, b0.z, b0.w, b1.x, b1.y, b1.z, b1.w};
        BF16Frag ahi, alo;
#pragma unroll
        for (int j = 0; j < 16; ++j) {
            unsigned bits = __float_as_uint(af[j]);
            ahi.h[j] = (unsigned short)(bits >> 16);
            float lo = af[j] - __uint_as_float(bits & 0xFFFF0000u);
            alo.h[j] = (unsigned short)(__float_as_uint(lo) >> 16);
        }
#pragma unroll
        for (int t = 0; t < 4; ++t) {
            BF16Frag bhi, blo;
#pragma unroll
            for (int jp = 0; jp < 4; ++jp) {
                uint2 x = *(const uint2*)&ldsB[0][ki][t][jp][lane][0];
                bhi.u[2 * jp]     = x.x;
                bhi.u[2 * jp + 1] = x.y;
                uint2 y = *(const uint2*)&ldsB[1][ki][t][jp][lane][0];
                blo.u[2 * jp]     = y.x;
                blo.u[2 * jp + 1] = y.y;
            }
            // fp32-accurate product via 3-term split-bf16
            acc[t] = __builtin_amdgcn_wmma_f32_16x16x32_bf16(
                false, ahi.v, false, bhi.v, (short)0, acc[t], false, false);
            acc[t] = __builtin_amdgcn_wmma_f32_16x16x32_bf16(
                false, ahi.v, false, blo.v, (short)0, acc[t], false, false);
            acc[t] = __builtin_amdgcn_wmma_f32_16x16x32_bf16(
                false, alo.v, false, bhi.v, (short)0, acc[t], false, false);
        }
    }

    // Epilogue: bias + relu + weighted sum over hidden dim.
    // C layout: VGPR i -> row (i + 8*(lane>=16)), column = lane&15 within tile.
    float qp[8] = {0.f, 0.f, 0.f, 0.f, 0.f, 0.f, 0.f, 0.f};
#pragma unroll
    for (int t = 0; t < 4; ++t) {
        int nh = t * 16 + (lane & 15);
        float bb = bemb[nh];
        float ws = Wedge[nh] + Wedge[HID + nh];      // W1 + W2 folded
#pragma unroll
        for (int i = 0; i < 8; ++i) {
            float h = acc[t][i] + bb;
            h = fmaxf(h, 0.0f);
            qp[i] = fmaf(h, ws, qp[i]);
        }
    }
    // Reduce across the 16 lanes that share each row (masks 1,2,4,8 stay in-half)
#pragma unroll
    for (int i = 0; i < 8; ++i) {
        float v = qp[i];
        v += __shfl_xor(v, 1);
        v += __shfl_xor(v, 2);
        v += __shfl_xor(v, 4);
        v += __shfl_xor(v, 8);
        qp[i] = v;
    }
    if ((lane & 15) == 0) {
        int mbase = tile * 16 + ((lane >> 4) << 3);
#pragma unroll
        for (int i = 0; i < 8; ++i) q[mbase + i] = qp[i];
    }
}

// ---------------------------------------------------------------------------
// Kernel: initialize degree arrays with the self-loop contribution (1 + EOS)
// ---------------------------------------------------------------------------
__global__ void deg_init_kernel(float* __restrict__ degs)
{
    int i = blockIdx.x * blockDim.x + threadIdx.x;
    if (i < 4 * N_NODES) degs[i] = 1.0f + EOSV;
}

// ---------------------------------------------------------------------------
// Kernel B: gumbel-sigmoid gates + degree scatter-adds (L2-resident atomics)
// ---------------------------------------------------------------------------
__launch_bounds__(256)
__global__ void edge_gate_kernel(const int* __restrict__ src,
                                 const int* __restrict__ dst,
                                 const float* __restrict__ noise,
                                 const float* __restrict__ q,
                                 const float* __restrict__ bedge,
                                 float* __restrict__ wlp_out,
                                 float* __restrict__ whp_out,
                                 float* __restrict__ degOutLp,
                                 float* __restrict__ degInLp,
                                 float* __restrict__ degOutHp,
                                 float* __restrict__ degInHp)
{
    int e = blockIdx.x * blockDim.x + threadIdx.x;
    if (e >= N_EDGES) return;
    int s = src[e];
    int d = dst[e];
    float raw = 0.5f * (q[s] + q[d]) + bedge[0];     // symmetrized edge score
    float u = noise[e];
    float eps = (2.0f * BIASV - 1.0f) * u + (1.0f - BIASV);
    float gate = logf(eps) - log1pf(-eps);
    float x = gate + raw;                            // TEMP = 1
    float wlp = 1.0f / (1.0f + expf(-x));
    float whp = 1.0f - wlp;
    wlp_out[e] = wlp;
    whp_out[e] = whp;
    float wlpe = wlp + EOSV;
    float whpe = whp + EOSV;
    atomicAdd(&degOutLp[s], wlpe);
    atomicAdd(&degInLp[d],  wlpe);
    atomicAdd(&degOutHp[s], whpe);
    atomicAdd(&degInHp[d],  whpe);
}

// ---------------------------------------------------------------------------
// Kernel C: EdgeWeightNorm('both') normalization + hp scaling / self-loop fixup
// ---------------------------------------------------------------------------
__launch_bounds__(256)
__global__ void edge_norm_kernel(const int* __restrict__ src,
                                 const int* __restrict__ dst,
                                 const float* __restrict__ wlp_raw,
                                 const float* __restrict__ degOutLp,
                                 const float* __restrict__ degInLp,
                                 const float* __restrict__ degOutHp,
                                 const float* __restrict__ degInHp,
                                 float* __restrict__ lp_out,
                                 float* __restrict__ hp_out)
{
    int i = blockIdx.x * blockDim.x + threadIdx.x;
    if (i >= N_EDGES + N_NODES) return;
    float lp, hp;
    if (i < N_EDGES) {
        int s = src[i];
        int d = dst[i];
        float wlp = wlp_raw[i];
        lp = (wlp + EOSV) / sqrtf(degOutLp[s] * degInLp[d]);
        hp = -ALPHAV * ((1.0f - wlp) + EOSV) / sqrtf(degOutHp[s] * degInHp[d]);
    } else {
        int n = i - N_EDGES;
        lp = (1.0f + EOSV) / sqrtf(degOutLp[n] * degInLp[n]);
        hp = 1.0f;                                   // self-loop forced to 1
    }
    lp_out[i] = lp;
    hp_out[i] = hp;
}

// ---------------------------------------------------------------------------
extern "C" void kernel_launch(void* const* d_in, const int* in_sizes, int n_in,
                              void* d_out, int out_size, void* d_ws, size_t ws_size,
                              hipStream_t stream)
{
    const float* feat  = (const float*)d_in[0];
    const int*   edges = (const int*)  d_in[1];
    const float* noise = (const float*)d_in[2];
    const float* Wemb  = (const float*)d_in[3];
    const float* bemb  = (const float*)d_in[4];
    const float* Wedge = (const float*)d_in[5];
    const float* bedge = (const float*)d_in[6];

    const int* src = edges;
    const int* dst = edges + N_EDGES;

    const int ENp = N_EDGES + N_NODES;
    float* out     = (float*)d_out;
    float* lp_out  = out;                        // w_lp_norm  [E+N]
    float* hp_out  = out + ENp;                  // w_hp_norm  [E+N]
    float* wlp_out = out + 2 * ENp;              // weights_lp [E]
    float* whp_out = out + 2 * ENp + N_EDGES;    // weights_hp [E]

    float* q        = (float*)d_ws;              // [N]
    float* degs     = q + N_NODES;               // 4 x [N]
    float* degOutLp = degs;
    float* degInLp  = degs + N_NODES;
    float* degOutHp = degs + 2 * N_NODES;
    float* degInHp  = degs + 3 * N_NODES;

    // 1) fused embedding GEMM -> q  (6250 tiles of 16 nodes, 8 waves/block)
    const int tiles = (N_NODES + 15) / 16;       // 6250
    embed_gemm_kernel<<<(tiles + 7) / 8, 256, 0, stream>>>(feat, Wemb, bemb, Wedge, q);

    // 2) degree init (self-loop contribution)
    deg_init_kernel<<<(4 * N_NODES + 255) / 256, 256, 0, stream>>>(degs);

    // 3) gates + degree accumulation
    edge_gate_kernel<<<(N_EDGES + 255) / 256, 256, 0, stream>>>(
        src, dst, noise, q, bedge, wlp_out, whp_out,
        degOutLp, degInLp, degOutHp, degInHp);

    // 4) normalization (reads finalized degrees; stream-ordered)
    edge_norm_kernel<<<(ENp + 255) / 256, 256, 0, stream>>>(
        src, dst, wlp_out, degOutLp, degInLp, degOutHp, degInHp, lp_out, hp_out);
}
